// EnergyConditionedFieldAttention_80805514707455
// MI455X (gfx1250) — matrix operand order repaired and checked by reading, not compiled, not run on hardware
//
#include <hip/hip_runtime.h>
#include <hip/hip_bf16.h>

// ---------------------------------------------------------------------------
// CDNA5 (gfx1250) WMMA implementation of EnergyConditionedFieldAttention.
// bf16 end-to-end datapath: weights pre-transposed to bf16, activations kept
// in bf16 between layers, f32 WMMA accumulation. Tiles staged with
// GLOBAL_LOAD_ASYNC_TO_LDS_B128 (ASYNCcnt) -- no VGPR round trip.
// 256-thread blocks (8 waves), 64x128 block tile, 32x32 per-wave tile,
// double-buffered LDS, b128 fragment loads -> v_wmma_f32_16x16x32_bf16.
// ---------------------------------------------------------------------------

typedef __attribute__((ext_vector_type(16))) __bf16 v16bf;
typedef __attribute__((ext_vector_type(8)))  __bf16 v8bf;
typedef __attribute__((ext_vector_type(4)))  __bf16 v4bf;
typedef __attribute__((ext_vector_type(8)))  float  v8f;

#define TILE_M 64
#define TILE_N 128
#define TILE_K 32
#define LDS_STRIDE 40   // 32 + 8 pad: 80B row stride, 16B aligned, low conflicts

__device__ __forceinline__ float silu_f(float x) {
    return x / (1.0f + __expf(-x));
}

__device__ __forceinline__ v16bf cat16(v8bf lo, v8bf hi) {
    return __builtin_shufflevector(lo, hi, 0, 1, 2, 3, 4, 5, 6, 7,
                                           8, 9, 10, 11, 12, 13, 14, 15);
}

__device__ __forceinline__ v4bf pack4(float4 v) {
    v4bf p;
    p[0] = (__bf16)v.x; p[1] = (__bf16)v.y;
    p[2] = (__bf16)v.z; p[3] = (__bf16)v.w;
    return p;
}

// Async copy of one 16-byte chunk, global -> LDS. ldsOff is the LDS byte
// offset (low 32 bits of the generic address of a __shared__ element).
__device__ __forceinline__ void async_copy_b128(unsigned int ldsOff,
                                                const __bf16* g) {
    asm volatile("global_load_async_to_lds_b128 %0, %1, off"
                 :: "v"(ldsOff), "v"(g) : "memory");
}

__device__ __forceinline__ void wait_async() {
    asm volatile("s_wait_asynccnt 0" ::: "memory");
}

// ---------------------------------------------------------------------------
// Uniform GEMM: out = act( A (MxK, bf16 row-major) * B (KxN) + bias ).
// B is stored pre-transposed: element (k,n) at Bt_g[n*ldb + k] (bf16).
// Outputs (any subset non-null):
//   Cf  : f32  row-major [M x N], ldc       (batch stride sCb)
//   Ch  : bf16 row-major [M x N], ldc       (batch stride sCb)
//   ChT : bf16 transposed [N x Mtot], ldcT  (no batch stride; used once)
// ---------------------------------------------------------------------------
__global__ __launch_bounds__(256)
void wmma_gemm_bf16(const __bf16* __restrict__ A, long long lda, long long sAb,
                    const __bf16* __restrict__ Bt_g, long long ldb, long long sBb,
                    const float* __restrict__ bias,
                    float* __restrict__ Cf, __bf16* __restrict__ Ch,
                    __bf16* __restrict__ ChT,
                    long long ldc, long long ldcT, long long sCb,
                    int M, int N, int K, int act)
{
    __shared__ __bf16 As[2][TILE_M][LDS_STRIDE];   // A tile [m][k]
    __shared__ __bf16 Bs[2][TILE_N][LDS_STRIDE];   // B tile [n][k]

    const int tid  = threadIdx.x;
    const int lane = tid & 31;
    const int wid  = tid >> 5;
    const int wm   = wid & 1;     // 2 row groups of 32
    const int wn   = wid >> 1;    // 4 col groups of 32

    const int rowBase = blockIdx.y * TILE_M;
    const int colBase = blockIdx.x * TILE_N;
    const bool fullM  = (rowBase + TILE_M) <= M;

    const __bf16* Ab = A    + (long long)blockIdx.z * sAb;
    const __bf16* Bb = Bt_g + (long long)blockIdx.z * sBb;

    const int half = lane >> 4;   // 0: lanes 0-15, 1: lanes 16-31
    const int l15  = lane & 15;

    v8f c00 = {}, c01 = {}, c10 = {}, c11 = {};

    // ---- async tile staging: pure byte copy global -> LDS -----------------
    auto stage = [&](int kt, int bufi) {
        // A tile: 64 rows x 64B = 256 x b128, one per thread.
        {
            int r  = tid >> 2;
            int k8 = (tid & 3) << 3;
            int gr = rowBase + r;
            if (gr >= M) gr = M - 1;   // clamp: rows >= M never stored anyway
            async_copy_b128((unsigned int)(size_t)&As[bufi][r][k8],
                            Ab + (long long)gr * lda + (kt + k8));
        }
        // B tile: 128 rows x 64B = 512 x b128, two per thread.
        #pragma unroll
        for (int it = 0; it < 2; ++it) {
            int idx = tid + it * 256;
            int nn  = idx >> 2;
            int k8  = (idx & 3) << 3;
            async_copy_b128((unsigned int)(size_t)&Bs[bufi][nn][k8],
                            Bb + (long long)(colBase + nn) * ldb + (kt + k8));
        }
    };

    // ---- fragment math: 2 A-frags x 2 B-frags -> 4 WMMAs ------------------
    auto compute = [&](int bufi) {
        // A frag: K = e + 16*half (one contiguous 16-run per lane, matches B)
        const __bf16* a0p = &As[bufi][wm * 32 + l15][0];
        const __bf16* a1p = &As[bufi][wm * 32 + 16 + l15][0];
        v16bf a0 = cat16(*(const v8bf*)(a0p + 16 * half),
                         *(const v8bf*)(a0p + 16 * half + 8));
        v16bf a1 = cat16(*(const v8bf*)(a1p + 16 * half),
                         *(const v8bf*)(a1p + 16 * half + 8));
        // B frag: K = e + 16*half (one contiguous 16-run per lane)
        const __bf16* b0p = &Bs[bufi][wn * 32 + l15][16 * half];
        const __bf16* b1p = &Bs[bufi][wn * 32 + 16 + l15][16 * half];
        v16bf b0 = cat16(*(const v8bf*)(b0p), *(const v8bf*)(b0p + 8));
        v16bf b1 = cat16(*(const v8bf*)(b1p), *(const v8bf*)(b1p + 8));

        c00 = __builtin_amdgcn_wmma_f32_16x16x32_bf16(false, a0, false, b0,
                                                      (short)0, c00, false, false);
        c01 = __builtin_amdgcn_wmma_f32_16x16x32_bf16(false, a0, false, b1,
                                                      (short)0, c01, false, false);
        c10 = __builtin_amdgcn_wmma_f32_16x16x32_bf16(false, a1, false, b0,
                                                      (short)0, c10, false, false);
        c11 = __builtin_amdgcn_wmma_f32_16x16x32_bf16(false, a1, false, b1,
                                                      (short)0, c11, false, false);
    };

    // ---- double-buffered main loop ----------------------------------------
    stage(0, 0);
    wait_async();
    __syncthreads();
    int buf = 0;
    for (int kt = 0; kt < K; kt += TILE_K) {
        if (kt + TILE_K < K)
            stage(kt + TILE_K, buf ^ 1);   // async DMA overlaps the WMMAs
        compute(buf);
        wait_async();
        __syncthreads();
        buf ^= 1;
    }

    // ---- epilogue: bias + optional SiLU, multi-format stores --------------
    // C/D layout: VGPR j -> M = j + 8*half ; N = l15
    float*  Cfb = Cf ? Cf + (long long)blockIdx.z * sCb : nullptr;
    __bf16* Chb = Ch ? Ch + (long long)blockIdx.z * sCb : nullptr;
    const int mHi = half * 8;
    auto epilogue = [&](bool guard) {
        #pragma unroll
        for (int i = 0; i < 2; ++i) {
            #pragma unroll
            for (int j = 0; j < 8; ++j) {
                int m = rowBase + wm * 32 + i * 16 + j + mHi;
                if (!guard || m < M) {
                    int   n0 = colBase + wn * 32 + l15;
                    int   n1 = n0 + 16;
                    float x0 = (i == 0 ? c00[j] : c10[j]) + (bias ? bias[n0] : 0.0f);
                    float x1 = (i == 0 ? c01[j] : c11[j]) + (bias ? bias[n1] : 0.0f);
                    if (act) { x0 = silu_f(x0); x1 = silu_f(x1); }
                    if (Cfb) {
                        Cfb[(long long)m * ldc + n0] = x0;
                        Cfb[(long long)m * ldc + n1] = x1;
                    }
                    if (Chb) {
                        Chb[(long long)m * ldc + n0] = (__bf16)x0;
                        Chb[(long long)m * ldc + n1] = (__bf16)x1;
                    }
                    if (ChT) {
                        ChT[(long long)n0 * ldcT + m] = (__bf16)x0;
                        ChT[(long long)n1 * ldcT + m] = (__bf16)x1;
                    }
                }
            }
        }
    };
    if (fullM) epilogue(false);
    else       epilogue(true);
}

// ---- elementwise f32 -> bf16 (4 per thread) -------------------------------
__global__ __launch_bounds__(256)
void cvt_f32_bf16(const float* __restrict__ x, __bf16* __restrict__ y,
                  long long n)
{
    long long i = ((long long)blockIdx.x * 256 + threadIdx.x) * 4;
    if (i + 3 < n) {
        float4 v = *(const float4*)(x + i);
        *(v4bf*)(y + i) = pack4(v);
    } else {
        for (long long j = i; j < n; ++j) y[j] = (__bf16)x[j];
    }
}

// ---- weight transpose + convert: W[KxN] f32 -> Wt[NxK] bf16 ---------------
__global__ __launch_bounds__(256)
void transpose_f32_bf16(const float* __restrict__ W, __bf16* __restrict__ Wt,
                        int K, int N)
{
    long long o = (long long)blockIdx.x * 256 + threadIdx.x;   // write-coalesced
    if (o < (long long)K * N) {
        int n = (int)(o / K);
        int k = (int)(o % K);
        Wt[o] = (__bf16)W[(long long)k * N + n];
    }
}

// Fused scale + mask + softmax(axis=N) + post-softmax mask.
// Reads f32 scores, writes bf16 attention weights (A of attn @ V).
__global__ __launch_bounds__(256)
void softmax_mask_kernel(const float* __restrict__ scores,
                         __bf16* __restrict__ attn,
                         const unsigned char* __restrict__ mask,
                         int E, int N, float scale)
{
    const int e = blockIdx.x;
    const int b = blockIdx.y;
    const float* row = scores + ((long long)b * E + e) * N;
    __bf16*      out = attn   + ((long long)b * E + e) * N;
    const unsigned char* mrow = mask + (long long)b * N;

    __shared__ float red[256];
    const int tid = threadIdx.x;
    const int n0 = tid, n1 = tid + 256;

    const bool m0 = mrow[n0] != 0;
    const bool m1 = mrow[n1] != 0;
    float v0 = m0 ? row[n0] * scale : -1e9f;
    float v1 = m1 ? row[n1] * scale : -1e9f;

    red[tid] = fmaxf(v0, v1);
    __syncthreads();
    #pragma unroll
    for (int s = 128; s > 0; s >>= 1) {
        if (tid < s) red[tid] = fmaxf(red[tid], red[tid + s]);
        __syncthreads();
    }
    const float mx = red[0];
    __syncthreads();

    float e0 = __expf(v0 - mx);
    float e1 = __expf(v1 - mx);
    red[tid] = e0 + e1;
    __syncthreads();
    #pragma unroll
    for (int s = 128; s > 0; s >>= 1) {
        if (tid < s) red[tid] += red[tid + s];
        __syncthreads();
    }
    const float inv = 1.0f / red[0];

    out[n0] = (__bf16)(m0 ? e0 * inv : 0.0f);   // post-softmax mask
    out[n1] = (__bf16)(m1 ? e1 * inv : 0.0f);
}

// ---------------------------------------------------------------------------

extern "C" void kernel_launch(void* const* d_in, const int* in_sizes, int n_in,
                              void* d_out, int out_size, void* d_ws, size_t ws_size,
                              hipStream_t stream)
{
    (void)in_sizes; (void)n_in; (void)out_size; (void)ws_size;

    constexpr long long Bsz = 64, Nn = 512, NE = 500;
    constexpr long long FD = 256, ED = 64, HID = 512, LAT = 256;
    constexpr long long MT = Bsz * Nn;    // 32768 field rows

    const float*         field  = (const float*)d_in[0];          // [B,N,FD]
    const unsigned char* mask   = (const unsigned char*)d_in[1];  // [B,N] bool
    const float*         e_feat = (const float*)d_in[2];          // [NE,ED]
    const float* q_w1 = (const float*)d_in[3];  const float* q_b1 = (const float*)d_in[4];
    const float* q_w2 = (const float*)d_in[5];  const float* q_b2 = (const float*)d_in[6];
    const float* q_w3 = (const float*)d_in[7];  const float* q_b3 = (const float*)d_in[8];
    const float* k_w1 = (const float*)d_in[9];  const float* k_b1 = (const float*)d_in[10];
    const float* k_w2 = (const float*)d_in[11]; const float* k_b2 = (const float*)d_in[12];
    const float* k_w3 = (const float*)d_in[13]; const float* k_b3 = (const float*)d_in[14];
    const float* v_w1 = (const float*)d_in[15]; const float* v_b1 = (const float*)d_in[16];
    const float* v_w2 = (const float*)d_in[17]; const float* v_b2 = (const float*)d_in[18];
    const float* v_w3 = (const float*)d_in[19]; const float* v_b3 = (const float*)d_in[20];
    const float* o_w1 = (const float*)d_in[21]; const float* o_b1 = (const float*)d_in[22];
    const float* o_w2 = (const float*)d_in[23]; const float* o_b2 = (const float*)d_in[24];

    // ---- workspace layout (256B-aligned buffers) --------------------------
    char* base = (char*)d_ws;
    size_t off = 0;
    auto allocB = [&](size_t bytes) {
        off = (off + 255) & ~(size_t)255;
        void* p = base + off;
        off += bytes;
        return p;
    };
    auto abf = [&](long long n) { return (__bf16*)allocB((size_t)n * 2); };
    auto af32 = [&](long long n) { return (float*)allocB((size_t)n * 4); };

    __bf16* fieldh  = abf(MT * FD);        // bf16 field
    __bf16* e_feath = abf(NE * ED);
    __bf16* wq1t = abf(ED * HID);  __bf16* wq2t = abf(HID * HID);
    __bf16* wq3t = abf(HID * LAT);
    __bf16* wk1t = abf(FD * HID);  __bf16* wk2t = abf(HID * HID);
    __bf16* wk3t = abf(HID * LAT);
    __bf16* wv1t = abf(FD * HID);  __bf16* wv2t = abf(HID * HID);
    __bf16* wv3t = abf(HID * LAT);
    __bf16* wo1t = abf(LAT * HID); __bf16* wo2t = abf(HID * LAT);
    __bf16* qh1  = abf(NE * HID);
    __bf16* qh2  = abf(NE * HID);
    __bf16* qb   = abf(NE * LAT);
    __bf16* h1   = abf(MT * HID);          // hidden buffer A
    __bf16* h2   = abf(MT * HID);          // hidden buffer B
    __bf16* kb16 = abf(MT * LAT);          // K latents [key][lat] == B^T layout
    __bf16* vt16 = abf(LAT * MT);          // V latents TRANSPOSED [lat][b*512+key]
    float*  sc   = af32(Bsz * NE * Nn);    // scores f32
    __bf16* oh16 = abf(Bsz * NE * HID);    // out-MLP hidden
    __bf16* attn16 = h2;                   // attn weights (reuses h2, free by then)
    __bf16* ao16   = h1;                   // attn output  (reuses h1, free by then)

    // ---- conversions ------------------------------------------------------
    auto cvt = [&](const float* x, __bf16* y, long long n) {
        cvt_f32_bf16<<<(unsigned)((n + 1023) / 1024), 256, 0, stream>>>(x, y, n);
    };
    auto wt = [&](const float* W, __bf16* Wt, long long K, long long N) {
        transpose_f32_bf16<<<(unsigned)((K * N + 255) / 256), 256, 0, stream>>>(
            W, Wt, (int)K, (int)N);
    };
    cvt(field, fieldh, MT * FD);
    cvt(e_feat, e_feath, NE * ED);
    wt(q_w1, wq1t, ED, HID);  wt(q_w2, wq2t, HID, HID);  wt(q_w3, wq3t, HID, LAT);
    wt(k_w1, wk1t, FD, HID);  wt(k_w2, wk2t, HID, HID);  wt(k_w3, wk3t, HID, LAT);
    wt(v_w1, wv1t, FD, HID);  wt(v_w2, wv2t, HID, HID);  wt(v_w3, wv3t, HID, LAT);
    wt(o_w1, wo1t, LAT, HID); wt(o_w2, wo2t, HID, LAT);

    // ---- GEMM launcher ----------------------------------------------------
    auto gemm = [&](const __bf16* A, long long lda, long long sAb,
                    const __bf16* Bt, long long ldb, long long sBb,
                    const float* bias,
                    float* Cf, __bf16* Ch, __bf16* ChT,
                    long long ldc, long long ldcT, long long sCb,
                    long long M, long long N, long long K, int act,
                    long long batch) {
        dim3 grid((unsigned)(N / TILE_N), (unsigned)((M + TILE_M - 1) / TILE_M),
                  (unsigned)batch);
        wmma_gemm_bf16<<<grid, 256, 0, stream>>>(A, lda, sAb, Bt, ldb, sBb, bias,
                                                 Cf, Ch, ChT, ldc, ldcT, sCb,
                                                 (int)M, (int)N, (int)K, act);
    };

    // ---- Q = mlp3(e_feat) -------------------------------------------------
    gemm(e_feath, ED, 0,  wq1t, ED, 0,  q_b1, nullptr, qh1, nullptr,
         HID, 0, 0,  NE, HID, ED, 1, 1);
    gemm(qh1, HID, 0,     wq2t, HID, 0, q_b2, nullptr, qh2, nullptr,
         HID, 0, 0,  NE, HID, HID, 1, 1);
    gemm(qh2, HID, 0,     wq3t, HID, 0, q_b3, nullptr, qb, nullptr,
         LAT, 0, 0,  NE, LAT, HID, 0, 1);

    // ---- K = mlp3(field); latents land row-major (= B^T layout for Q.K^T) -
    gemm(fieldh, FD, 0,   wk1t, FD, 0,  k_b1, nullptr, h1, nullptr,
         HID, 0, 0,  MT, HID, FD, 1, 1);
    gemm(h1, HID, 0,      wk2t, HID, 0, k_b2, nullptr, h2, nullptr,
         HID, 0, 0,  MT, HID, HID, 1, 1);
    gemm(h2, HID, 0,      wk3t, HID, 0, k_b3, nullptr, kb16, nullptr,
         LAT, 0, 0,  MT, LAT, HID, 0, 1);

    // ---- V = mlp3(field); latents written fully TRANSPOSED ----------------
    gemm(fieldh, FD, 0,   wv1t, FD, 0,  v_b1, nullptr, h1, nullptr,
         HID, 0, 0,  MT, HID, FD, 1, 1);
    gemm(h1, HID, 0,      wv2t, HID, 0, v_b2, nullptr, h2, nullptr,
         HID, 0, 0,  MT, HID, HID, 1, 1);
    gemm(h2, HID, 0,      wv3t, HID, 0, v_b3, nullptr, nullptr, vt16,
         0, MT, 0,   MT, LAT, HID, 0, 1);

    // ---- scores[b] = Q (NExLAT) x K[b]^T : B = kb16, ldb=LAT, batch=512 rows
    gemm(qb, LAT, 0,
         kb16, LAT, Nn * LAT,
         nullptr, sc, nullptr, nullptr,
         Nn, 0, NE * Nn,
         NE, Nn, LAT, 0, Bsz);

    // ---- fused scale + mask + softmax + post-mask -> bf16 attn ------------
    softmax_mask_kernel<<<dim3((unsigned)NE, (unsigned)Bsz), 256, 0, stream>>>(
        sc, attn16, mask, (int)NE, (int)Nn, 0.0625f /* LAT^-0.5 */);

    // ---- attnout[b] = attn[b] (NExNn) x V[b] : B = vt16, ldb=MT, batch=512 -
    gemm(attn16, Nn, NE * Nn,
         vt16, MT, Nn,
         nullptr, nullptr, ao16, nullptr,
         LAT, 0, NE * LAT,
         NE, LAT, Nn, 0, Bsz);

    // ---- out = mlp2(attnout) : flat [B*NE, LAT] ---------------------------
    gemm(ao16, LAT, 0,  wo1t, LAT, 0,  o_b1, nullptr, oh16, nullptr,
         HID, 0, 0,  Bsz * NE, HID, LAT, 1, 1);
    gemm(oh16, HID, 0,  wo2t, HID, 0,  o_b2, (float*)d_out, nullptr, nullptr,
         LAT, 0, 0,  Bsz * NE, LAT, HID, 0, 1);
}
